// AttentionBlock3D_53137335386668
// MI455X (gfx1250) — matrix operand order repaired and checked by reading, not compiled, not run on hardware
//
#include <hip/hip_runtime.h>
#include <hip/hip_bf16.h>
#include <hip/hip_fp16.h>

typedef __attribute__((ext_vector_type(16))) _Float16 v16h;
typedef __attribute__((ext_vector_type(8)))  _Float16 v8h;
typedef __attribute__((ext_vector_type(8)))  float    v8f;
typedef int v4i __attribute__((vector_size(16)));

constexpr int B_  = 2;
constexpr int C_  = 256;
constexpr int N_  = 4096;   // 16*16*16
constexpr int NH_ = 4;
constexpr int HD_ = 64;
constexpr int G_  = 8;

// ---------------------------------------------------------------------------
// WMMA fragment helpers (wave32; A and B both use contiguous k = 16*(lane/16) + s)
// ---------------------------------------------------------------------------

// A fragment: 16x32 f16 tile, row-major, leading dim ldk (elements).
__device__ __forceinline__ v16h load_frag_a(const _Float16* base, int ldk) {
  int l = threadIdx.x & 31;
  const _Float16* p = base + (size_t)(l & 15) * ldk + ((l >> 4) << 4);
  v8h lo = *(const v8h*)p;
  v8h hi = *(const v8h*)(p + 8);
  v16h f;
#pragma unroll
  for (int i = 0; i < 8; ++i) { f[i] = lo[i]; f[i + 8] = hi[i]; }
  return f;
}

// B fragment: 32x16 (KxN) where storage is N-major (16 rows of length ldk, K-minor).
__device__ __forceinline__ v16h load_frag_b(const _Float16* base, int ldk) {
  int l = threadIdx.x & 31;
  const _Float16* p = base + (size_t)(l & 15) * ldk + ((l >> 4) << 4);
  return *(const v16h*)p;   // 32B, alignment guaranteed by layout choices
}

__device__ __forceinline__ v8f wmma16(v16h a, v16h b, v8f c) {
  return __builtin_amdgcn_wmma_f32_16x16x32_f16(false, a, false, b, (short)0, c,
                                                false, false);
}

// CDNA5 async global->LDS copy (ASYNCcnt-tracked), 16B per lane.
__device__ __forceinline__ void async_cp_b128(const void* g, void* l) {
  __builtin_amdgcn_global_load_async_to_lds_b128(
      (__attribute__((address_space(1))) v4i*)g,
      (__attribute__((address_space(3))) v4i*)l, 0, 0);
}

// ---------------------------------------------------------------------------
// 1) GroupNorm statistics: one block per (b, group); slab is contiguous.
// ---------------------------------------------------------------------------
__global__ void gn_stats_kernel(const float* __restrict__ x, float* __restrict__ stats) {
  const int slab = (C_ / G_) * N_;  // 32*4096 contiguous floats
  const float* p = x + (size_t)blockIdx.x * slab;
  float s = 0.f, ss = 0.f;
  for (int i = threadIdx.x; i < slab; i += blockDim.x) {
    float v = p[i];
    s += v; ss += v * v;
  }
  __shared__ float sh[512];
  sh[threadIdx.x] = s;
  sh[threadIdx.x + 256] = ss;
  __syncthreads();
  for (int off = 128; off > 0; off >>= 1) {
    if ((int)threadIdx.x < off) {
      sh[threadIdx.x] += sh[threadIdx.x + off];
      sh[threadIdx.x + 256] += sh[threadIdx.x + 256 + off];
    }
    __syncthreads();
  }
  if (threadIdx.x == 0) {
    float mean = sh[0] / (float)slab;
    float var  = sh[256] / (float)slab - mean * mean;
    stats[blockIdx.x * 2 + 0] = mean;
    stats[blockIdx.x * 2 + 1] = rsqrtf(var + 1e-5f);
  }
}

// ---------------------------------------------------------------------------
// 2) GroupNorm apply -> h16 in [b][n][c] layout (f16, K-minor for WMMA B frags)
// ---------------------------------------------------------------------------
__global__ void gn_apply_kernel(const float* __restrict__ x,
                                const float* __restrict__ nw,
                                const float* __restrict__ nb,
                                const float* __restrict__ stats,
                                _Float16* __restrict__ h16) {
  int b = blockIdx.x >> 8;      // blockIdx = b*C + c
  int c = blockIdx.x & 255;
  int g = c >> 5;               // 32 channels per group
  float mean = stats[(b * G_ + g) * 2 + 0];
  float rstd = stats[(b * G_ + g) * 2 + 1];
  float w  = nw[c] * rstd;
  float bb = nb[c] - mean * w;
  const float* xp = x + ((size_t)b * C_ + c) * N_;
  _Float16* hp = h16 + (size_t)b * N_ * C_ + c;
  for (int n = threadIdx.x; n < N_; n += blockDim.x)
    hp[(size_t)n * C_] = (_Float16)(xp[n] * w + bb);
}

// ---------------------------------------------------------------------------
// 3) Convert GEMM weights f32 -> f16
// ---------------------------------------------------------------------------
__global__ void cvt_w_kernel(const float* __restrict__ qkv_w,
                             const float* __restrict__ proj_w,
                             _Float16* __restrict__ qw16,
                             _Float16* __restrict__ pw16) {
  int i = blockIdx.x * blockDim.x + threadIdx.x;
  if (i < 768 * 256) qw16[i] = (_Float16)qkv_w[i];
  if (i < 256 * 256) pw16[i] = (_Float16)proj_w[i];
}

// ---------------------------------------------------------------------------
// 4) QKV GEMM: D[oc,n] = W[oc,:] * h[:,n] + b.  One wave per 16x16 tile.
//    Scatter Q,K -> [bh][n][64] (K-minor), V -> [bh][64][n] (m-minor).
//    Q pre-scaled by hd^-0.5 = 0.125.
// ---------------------------------------------------------------------------
__global__ void qkv_kernel(const _Float16* __restrict__ h16,
                           const _Float16* __restrict__ qw16,
                           const float* __restrict__ qkv_b,
                           _Float16* __restrict__ q16,
                           _Float16* __restrict__ k16,
                           _Float16* __restrict__ v16) {
  int wave = blockIdx.x * (blockDim.x >> 5) + (threadIdx.x >> 5);
  int nT  = wave & 255;             // N/16 tiles
  int t   = wave >> 8;
  int ocT = t % 48;                 // 768/16
  int b   = t / 48;

  const _Float16* abase = qw16 + (size_t)ocT * 16 * C_;
  const _Float16* bbase = h16 + ((size_t)b * N_ + (size_t)nT * 16) * C_;
  v8f acc = {};
#pragma unroll
  for (int kk = 0; kk < 8; ++kk) {
    v16h A  = load_frag_a(abase + kk * 32, C_);
    v16h Bf = load_frag_b(bbase + kk * 32, C_);
    acc = wmma16(A, Bf, acc);
  }

  int l = threadIdx.x & 31;
  int col = l & 15;
  int rowbase = (l >> 4) << 3;       // 0 or 8 (D-tile row half)
  int oc0 = ocT * 16;
  int n = nT * 16 + col;
  int mtype = oc0 >> 8;              // 0=Q 1=K 2=V (tile never crosses)
  int head = (oc0 & 255) >> 6;
  int cin  = oc0 & 63;               // multiple of 16

  if (mtype == 2) {
    _Float16* vp = v16 + (((size_t)(b * NH_ + head)) * HD_ + cin + rowbase) * (size_t)N_ + n;
#pragma unroll
    for (int r = 0; r < 8; ++r)
      vp[(size_t)r * N_] = (_Float16)(acc[r] + qkv_b[oc0 + rowbase + r]);
  } else {
    float scale = (mtype == 0) ? 0.125f : 1.0f;
    _Float16* qp = ((mtype == 0) ? q16 : k16)
                 + (((size_t)(b * NH_ + head)) * N_ + n) * HD_ + cin + rowbase;
    v8h pack;
#pragma unroll
    for (int r = 0; r < 8; ++r)
      pack[r] = (_Float16)((acc[r] + qkv_b[oc0 + rowbase + r]) * scale);
    *(v8h*)qp = pack;               // 8 consecutive channels, 16B aligned
  }
}

// ---------------------------------------------------------------------------
// 5) Flash attention, LDS-staged K/V shared by the whole block.
//    Block = 4 waves, all on the same (b,head); each wave owns 16 query rows.
//    K/V tiles (32 keys) are double-buffered in LDS via async global->LDS
//    copies (ASYNCcnt), overlapped with the 8 WMMAs + online softmax.
// ---------------------------------------------------------------------------
__global__ void attn_kernel(const _Float16* __restrict__ q16,
                            const _Float16* __restrict__ k16,
                            const _Float16* __restrict__ v16,
                            _Float16* __restrict__ o16) {
  __shared__ _Float16 ldsK[2][32 * 64];   // [buf][m 0..31][c 0..63]   4KB each
  __shared__ _Float16 ldsV[2][64 * 32];   // [buf][c 0..63][m 0..31]   4KB each
  __shared__ _Float16 ldsP[4][16 * 32];   // wave-private P tiles

  int wave = threadIdx.x >> 5;
  int l = threadIdx.x & 31;
  int tid = threadIdx.x;

  int bh = blockIdx.x >> 6;                  // 64 blocks per (b,head)
  int nq = ((blockIdx.x & 63) * 4 + wave) * 16;
  int b = bh >> 2, head = bh & 3;

  const _Float16* kb = k16 + (size_t)bh * N_ * HD_;  // [m][64]
  const _Float16* vb = v16 + (size_t)bh * HD_ * N_;  // [c][4096]

  // Q fragments (K = c 0..31 / 32..63)
  const _Float16* qbase = q16 + ((size_t)bh * N_ + nq) * HD_;
  v16h qa0 = load_frag_a(qbase, HD_);
  v16h qa1 = load_frag_a(qbase + 32, HD_);

  v8f o0 = {}, o1 = {}, o2 = {}, o3 = {};
  float mrow[8], lrow[8];
#pragma unroll
  for (int r = 0; r < 8; ++r) { mrow[r] = -3.0e38f; lrow[r] = 0.f; }

  _Float16* myP = ldsP[wave];
  int rowoff = (l >> 4) << 3;
  int c0 = l & 15;

  // --- async tile loader: all 128 threads cooperatively stage one 32-key tile
  auto issue_tile = [&](int buf, int m0) {
    // K tile: contiguous 32*64 f16 = 4KB -> 256 x 16B chunks
    const char* gk = (const char*)(kb + (size_t)m0 * HD_);
    char* lk = (char*)&ldsK[buf][0];
    async_cp_b128(gk + tid * 16, lk + tid * 16);
    async_cp_b128(gk + (tid + 128) * 16, lk + (tid + 128) * 16);
    // V tile: 64 rows x 64B (stride N_*2) -> 256 x 16B chunks
#pragma unroll
    for (int j = tid; j < 256; j += 128) {
      int c = j >> 2, mo = (j & 3) * 8;
      async_cp_b128(vb + (size_t)c * N_ + m0 + mo, &ldsV[buf][c * 32 + mo]);
    }
  };

  issue_tile(0, 0);

  for (int it = 0; it < N_ / 32; ++it) {
    int buf = it & 1;
    __builtin_amdgcn_s_wait_asynccnt(0);   // this wave's copies landed
    __syncthreads();                       // everyone's copies landed
    if (it + 1 < N_ / 32) issue_tile(buf ^ 1, (it + 1) * 32);

    const _Float16* Kt = &ldsK[buf][0];
    const _Float16* Vt = &ldsV[buf][0];

    // ---- S = Q K^T for 32 key columns (two 16-wide subtiles)
    v8f s0 = {}, s1 = {};
    {
      v16h kb0 = load_frag_b(Kt, 64);
      v16h kb1 = load_frag_b(Kt + 32, 64);
      s0 = wmma16(qa0, kb0, s0);
      s0 = wmma16(qa1, kb1, s0);
      v16h kb2 = load_frag_b(Kt + 16 * 64, 64);
      v16h kb3 = load_frag_b(Kt + 16 * 64 + 32, 64);
      s1 = wmma16(qa0, kb2, s1);
      s1 = wmma16(qa1, kb3, s1);
    }
    // ---- online softmax (rows live across 16-lane halves; xor 1/2/4/8)
#pragma unroll
    for (int r = 0; r < 8; ++r) {
      float t = fmaxf(s0[r], s1[r]);
      t = fmaxf(t, __shfl_xor(t, 1, 32));
      t = fmaxf(t, __shfl_xor(t, 2, 32));
      t = fmaxf(t, __shfl_xor(t, 4, 32));
      t = fmaxf(t, __shfl_xor(t, 8, 32));
      float mnew = fmaxf(mrow[r], t);
      float alpha = __expf(mrow[r] - mnew);
      float p0 = __expf(s0[r] - mnew);
      float p1 = __expf(s1[r] - mnew);
      float rs = p0 + p1;
      rs += __shfl_xor(rs, 1, 32);
      rs += __shfl_xor(rs, 2, 32);
      rs += __shfl_xor(rs, 4, 32);
      rs += __shfl_xor(rs, 8, 32);
      lrow[r] = lrow[r] * alpha + rs;
      mrow[r] = mnew;
      o0[r] *= alpha; o1[r] *= alpha; o2[r] *= alpha; o3[r] *= alpha;
      s0[r] = p0; s1[r] = p1;
    }
    // ---- P (D-layout) -> LDS row-major [16][32] -> A-fragment
#pragma unroll
    for (int r = 0; r < 8; ++r) {
      myP[(rowoff + r) * 32 + c0]      = (_Float16)s0[r];
      myP[(rowoff + r) * 32 + c0 + 16] = (_Float16)s1[r];
    }
    v16h pa = load_frag_a(myP, 32);
    // ---- O += P * V^T over the 4 channel subtiles of hd=64
    v16h vb0 = load_frag_b(Vt + 0 * 16 * 32, 32);
    v16h vb1 = load_frag_b(Vt + 1 * 16 * 32, 32);
    v16h vb2 = load_frag_b(Vt + 2 * 16 * 32, 32);
    v16h vb3 = load_frag_b(Vt + 3 * 16 * 32, 32);
    o0 = wmma16(pa, vb0, o0);
    o1 = wmma16(pa, vb1, o1);
    o2 = wmma16(pa, vb2, o2);
    o3 = wmma16(pa, vb3, o3);

    __syncthreads();   // all waves done with buf before it is refilled
  }

  // ---- epilogue: normalize and store o16 in [b][n][c] layout
  _Float16* op = o16 + ((size_t)b * N_ + nq + rowoff) * C_ + head * HD_ + c0;
#pragma unroll
  for (int r = 0; r < 8; ++r) {
    float inv = 1.0f / lrow[r];
    op[(size_t)r * C_ + 0]  = (_Float16)(o0[r] * inv);
    op[(size_t)r * C_ + 16] = (_Float16)(o1[r] * inv);
    op[(size_t)r * C_ + 32] = (_Float16)(o2[r] * inv);
    op[(size_t)r * C_ + 48] = (_Float16)(o3[r] * inv);
  }
}

// ---------------------------------------------------------------------------
// 6) Output projection + bias + residual (f32 out)
// ---------------------------------------------------------------------------
__global__ void proj_kernel(const _Float16* __restrict__ o16,
                            const _Float16* __restrict__ pw16,
                            const float* __restrict__ proj_b,
                            const float* __restrict__ x,
                            float* __restrict__ out) {
  int wave = blockIdx.x * (blockDim.x >> 5) + (threadIdx.x >> 5);
  int nT  = wave & 255;
  int t   = wave >> 8;
  int ocT = t & 15;                 // 256/16
  int b   = t >> 4;
  const _Float16* abase = pw16 + (size_t)ocT * 16 * C_;
  const _Float16* bbase = o16 + ((size_t)b * N_ + (size_t)nT * 16) * C_;
  v8f acc = {};
#pragma unroll
  for (int kk = 0; kk < 8; ++kk)
    acc = wmma16(load_frag_a(abase + kk * 32, C_),
                 load_frag_b(bbase + kk * 32, C_), acc);
  int l = threadIdx.x & 31;
  int col = l & 15, rowoff = (l >> 4) << 3;
  int n = nT * 16 + col;
  int oc0 = ocT * 16 + rowoff;
  const float* xp = x + ((size_t)b * C_ + oc0) * N_ + n;
  float* op = out + ((size_t)b * C_ + oc0) * N_ + n;
#pragma unroll
  for (int r = 0; r < 8; ++r)
    op[(size_t)r * N_] = acc[r] + proj_b[oc0 + r] + xp[(size_t)r * N_];
}

// ---------------------------------------------------------------------------
extern "C" void kernel_launch(void* const* d_in, const int* in_sizes, int n_in,
                              void* d_out, int out_size, void* d_ws, size_t ws_size,
                              hipStream_t stream) {
  const float* x      = (const float*)d_in[0];
  const float* norm_w = (const float*)d_in[1];
  const float* norm_b = (const float*)d_in[2];
  const float* qkv_w  = (const float*)d_in[3];
  const float* qkv_b  = (const float*)d_in[4];
  const float* proj_w = (const float*)d_in[5];
  const float* proj_b = (const float*)d_in[6];

  char* ws = (char*)d_ws;
  size_t off = 0;
  auto alloc = [&](size_t bytes) -> void* {
    void* p = ws + off;
    off = (off + bytes + 255) & ~(size_t)255;
    return p;
  };
  _Float16* h16  = (_Float16*)alloc((size_t)B_ * N_ * C_ * 2);       // 4 MB
  _Float16* qw16 = (_Float16*)alloc((size_t)768 * 256 * 2);
  _Float16* pw16 = (_Float16*)alloc((size_t)256 * 256 * 2);
  _Float16* q16  = (_Float16*)alloc((size_t)B_ * NH_ * N_ * HD_ * 2); // 4 MB
  _Float16* k16  = (_Float16*)alloc((size_t)B_ * NH_ * N_ * HD_ * 2);
  _Float16* v16  = (_Float16*)alloc((size_t)B_ * NH_ * HD_ * N_ * 2);
  _Float16* o16  = (_Float16*)alloc((size_t)B_ * N_ * C_ * 2);
  float*    stats = (float*)alloc((size_t)B_ * G_ * 2 * sizeof(float));

  gn_stats_kernel<<<B_ * G_, 256, 0, stream>>>(x, stats);
  gn_apply_kernel<<<B_ * C_, 256, 0, stream>>>(x, norm_w, norm_b, stats, h16);
  cvt_w_kernel<<<768, 256, 0, stream>>>(qkv_w, proj_w, qw16, pw16);
  // B * (768/16) * (4096/16) tiles, 4 waves / 128-thread block
  qkv_kernel<<<(B_ * 48 * 256) / 4, 128, 0, stream>>>(h16, qw16, qkv_b, q16, k16, v16);
  // B * NH * 64 blocks; each block = 4 query tiles sharing K/V via async LDS
  attn_kernel<<<B_ * NH_ * 64, 128, 0, stream>>>(q16, k16, v16, o16);
  // B * (256/16) * (4096/16) tiles
  proj_kernel<<<(B_ * 16 * 256) / 4, 128, 0, stream>>>(o16, pw16, proj_b, x, (float*)d_out);
}
